// ContextQueryAttentionLayer_1546188226838
// MI455X (gfx1250) — matrix operand order repaired and checked by reading, not compiled, not run on hardware
//
#include <hip/hip_runtime.h>

typedef __attribute__((ext_vector_type(16))) _Float16 v16h;
typedef __attribute__((ext_vector_type(8)))  float    v8f;
typedef __attribute__((ext_vector_type(4)))  float    f4;

#define BB 32
#define NN 1024
#define MM 256
#define DD 128

__device__ __forceinline__ float wave_sum(float v) {
#pragma unroll
  for (int o = 16; o >= 1; o >>= 1) v += __shfl_xor(v, o, 32);
  return v;
}
__device__ __forceinline__ float wave_max(float v) {
#pragma unroll
  for (int o = 16; o >= 1; o >>= 1) v = fmaxf(v, __shfl_xor(v, o, 32));
  return v;
}

// One block per batch. blockDim = 256 (8 waves).
// Computes c2q_ws[b][4][D] and q2c_ws[b][4][D].
__global__ __launch_bounds__(256)
void cqa_phase1(const float* __restrict__ context,
                const float* __restrict__ query,
                const float* __restrict__ w,
                float* __restrict__ c2q_ws,
                float* __restrict__ q2c_ws)
{
  __shared__ float s_srow[4][MM];     // the 4 distinct score rows
  __shared__ float s_sc[4][MM];       // row softmax (soft_c rows)
  __shared__ float s_e[4][MM];        // column exp (soft_q numerators)
  __shared__ float s_u[MM];           // 1/(256*Z[j])
  __shared__ float s_csp[8][4][DD];   // per-wave partial Cs
  __shared__ float s_cs[4][DD];       // Cs[r,d] = sum_{k%4==r} ctx[k,d]
  __shared__ float s_g[16];           // G[r][r']

  const int b    = blockIdx.x;
  const int tid  = threadIdx.x;
  const int lane = tid & 31;
  const int wave = tid >> 5;
  const float* ctx = context + (size_t)b * NN * DD;
  const float* qry = query   + (size_t)b * MM * DD;
  const int dbase = lane * 4;

  const float4 wq4 = *(const float4*)(w + dbase);
  const float4 wc4 = *(const float4*)(w + DD + dbase);
  const float4 wm4 = *(const float4*)(w + 2 * DD + dbase);

  // ---- Step A: per-row trilinear scores + strided context sums ----
  float cs[4][4] = {};
  for (int nn2 = 0; nn2 < 32; ++nn2) {
#pragma unroll
    for (int u = 0; u < 4; ++u) {
      const int n = wave * 128 + nn2 * 4 + u;   // n%4 == u
      const int j = n & (MM - 1);
      const int r = n >> 8;
      const float4 c4 = *(const float4*)(ctx + n * DD + dbase);
      const float4 q4 = *(const float4*)(qry + j * DD + dbase);
      float v = c4.x * wc4.x + c4.y * wc4.y + c4.z * wc4.z + c4.w * wc4.w;      // cdot
      v += q4.x * wq4.x + q4.y * wq4.y + q4.z * wq4.z + q4.w * wq4.w;           // qdot
      v += c4.x * wm4.x * q4.x + c4.y * wm4.y * q4.y
         + c4.z * wm4.z * q4.z + c4.w * wm4.w * q4.w;                           // diag(A)
      v = wave_sum(v);
      if (lane == 0) s_srow[r][j] = v;
      cs[u][0] += c4.x; cs[u][1] += c4.y; cs[u][2] += c4.z; cs[u][3] += c4.w;
    }
  }
#pragma unroll
  for (int u = 0; u < 4; ++u)
    *(float4*)(&s_csp[wave][u][dbase]) = make_float4(cs[u][0], cs[u][1], cs[u][2], cs[u][3]);
  __syncthreads();

  // ---- Step B: row softmax (waves 0-3) and column softmax (waves 4-7) ----
  if (wave < 4) {
    const int r = wave;
    float vals[8];
    float mx = -3.0e38f;
#pragma unroll
    for (int k = 0; k < 8; ++k) { vals[k] = s_srow[r][lane + 32 * k]; mx = fmaxf(mx, vals[k]); }
    mx = wave_max(mx);
    float s = 0.0f;
#pragma unroll
    for (int k = 0; k < 8; ++k) { vals[k] = __expf(vals[k] - mx); s += vals[k]; }
    s = wave_sum(s);
    const float inv = 1.0f / s;
#pragma unroll
    for (int k = 0; k < 8; ++k) s_sc[r][lane + 32 * k] = vals[k] * inv;
  } else {
#pragma unroll
    for (int t = 0; t < 2; ++t) {
      const int j = (wave - 4) * 64 + t * 32 + lane;
      const float a0 = s_srow[0][j], a1 = s_srow[1][j], a2 = s_srow[2][j], a3 = s_srow[3][j];
      const float m = fmaxf(fmaxf(a0, a1), fmaxf(a2, a3));
      const float e0 = __expf(a0 - m), e1 = __expf(a1 - m), e2 = __expf(a2 - m), e3 = __expf(a3 - m);
      s_e[0][j] = e0; s_e[1][j] = e1; s_e[2][j] = e2; s_e[3][j] = e3;
      s_u[j] = 1.0f / (256.0f * (e0 + e1 + e2 + e3));
    }
  }
  __syncthreads();

  // ---- Step C: G[r][r'] and Cs reduce, then q2c_r = G @ Cs ----
#pragma unroll
  for (int t = 0; t < 2; ++t) {
    const int p = wave * 2 + t;
    const int r = p >> 2, rp = p & 3;
    float acc = 0.0f;
#pragma unroll
    for (int jj = 0; jj < 8; ++jj) {
      const int j = lane + 32 * jj;
      acc += s_sc[r][j] * s_u[j] * s_e[rp][j];
    }
    acc = wave_sum(acc);
    if (lane == 0) s_g[p] = acc;
  }
#pragma unroll
  for (int t = 0; t < 2; ++t) {
    const int e = tid + t * 256;
    const int r = e >> 7, d = e & (DD - 1);
    float s = 0.0f;
#pragma unroll
    for (int wv = 0; wv < 8; ++wv) s += s_csp[wv][r][d];
    s_cs[r][d] = s;
  }
  __syncthreads();
#pragma unroll
  for (int t = 0; t < 2; ++t) {
    const int e = tid + t * 256;
    const int r = e >> 7, d = e & (DD - 1);
    const float v = s_g[r * 4 + 0] * s_cs[0][d] + s_g[r * 4 + 1] * s_cs[1][d]
                  + s_g[r * 4 + 2] * s_cs[2][d] + s_g[r * 4 + 3] * s_cs[3][d];
    q2c_ws[(b * 4 + r) * DD + d] = v;
  }

  // ---- Step D: c2q_r = sc(4->16 x 256) @ query(256 x 128) via WMMA f16 ----
  // Each wave owns a 16-wide d-tile; K=256 in 8 fully-unrolled chunks of 32
  // so fragment loads/cvts of chunk k+1 co-execute with the WMMA of chunk k.
  {
    const int d0    = wave * 16;
    const int row   = lane & 15;
    const int khalf = lane >> 4;
    const int col   = d0 + row;     // N index = lane%16
    v8f acc = {};
#pragma unroll
    for (int kc = 0; kc < 8; ++kc) {
      const int k0 = kc * 32;
      v16h a, bm;
#pragma unroll
      for (int i = 0; i < 16; ++i) {     // A 16x32 f16 layout (ISA 7.12.2)
        const int K = k0 + (i >> 3) * 16 + khalf * 8 + (i & 7);
        const float x = s_sc[row & 3][K];
        a[i] = (_Float16)(row < 4 ? x : 0.0f);   // value-select: EXEC stays all-1s
      }
#pragma unroll
      for (int i = 0; i < 16; ++i) {     // B 32x16 f16 layout
        const int K = k0 + (i >> 3) * 16 + khalf * 8 + (i & 7);
        bm[i] = (_Float16)qry[K * DD + col];
      }
      acc = __builtin_amdgcn_wmma_f32_16x16x32_f16(
          false, a, false, bm, (short)0, acc, false, false);
    }
    if (lane < 16) {
#pragma unroll
      for (int v = 0; v < 4; ++v)        // D layout: VGPR v = row M=v on lanes 0-15
        c2q_ws[(b * 4 + v) * DD + d0 + lane] = acc[v];
    }
  }
}

// Streaming writer: out[b,i,:] = [ctx, c2q_r[i%4], ctx*c2q_r, ctx*q2c_r].
// grid = B * (N/8); 8 waves per block, one row per wave, 128-bit lanes.
// Non-temporal hints: context's last read and a write-once output stream —
// keep them out of WGP$/L2 working set. Uses clang ext_vector f4 (native
// vector) so the nontemporal builtins accept the pointers.
__global__ __launch_bounds__(256)
void cqa_phase2(const float* __restrict__ context,
                const float* __restrict__ c2q_ws,
                const float* __restrict__ q2c_ws,
                float* __restrict__ out)
{
  __shared__ __align__(16) float s_c2q[4][DD];
  __shared__ __align__(16) float s_q2c[4][DD];
  const int b    = blockIdx.x >> 7;          // 128 row-tiles per batch
  const int tile = blockIdx.x & 127;
  const int tid  = threadIdx.x;
  const int lane = tid & 31;
  const int wave = tid >> 5;

  {
    const int idx = (tid & 127) * 4;
    const float* src = (tid < 128) ? (c2q_ws + b * 4 * DD) : (q2c_ws + b * 4 * DD);
    float* dst = (tid < 128) ? &s_c2q[0][0] : &s_q2c[0][0];
    *(f4*)(dst + idx) = *(const f4*)(src + idx);
  }
  __syncthreads();

  const int row   = tile * 8 + wave;
  const int r     = row & 3;
  const int dbase = lane * 4;
  const f4 c4 = __builtin_nontemporal_load(
      (const f4*)(context + ((size_t)b * NN + row) * DD + dbase));
  const f4 a4 = *(const f4*)(&s_c2q[r][dbase]);
  const f4 b4 = *(const f4*)(&s_q2c[r][dbase]);
  float* o = out + ((size_t)b * NN + row) * (4 * DD);
  __builtin_nontemporal_store(c4,      (f4*)(o + dbase));
  __builtin_nontemporal_store(a4,      (f4*)(o + DD + dbase));
  __builtin_nontemporal_store(c4 * a4, (f4*)(o + 2 * DD + dbase));
  __builtin_nontemporal_store(c4 * b4, (f4*)(o + 3 * DD + dbase));
}

extern "C" void kernel_launch(void* const* d_in, const int* in_sizes, int n_in,
                              void* d_out, int out_size, void* d_ws, size_t ws_size,
                              hipStream_t stream) {
  (void)in_sizes; (void)n_in; (void)out_size; (void)ws_size;
  const float* context = (const float*)d_in[0];
  const float* query   = (const float*)d_in[1];
  // d_in[2], d_in[3] are all-true masks; unused by the reference math.
  const float* w       = (const float*)d_in[4];
  float* out    = (float*)d_out;
  float* c2q_ws = (float*)d_ws;                // [B][4][D]
  float* q2c_ws = c2q_ws + BB * 4 * DD;        // [B][4][D]  (128 KB total)

  cqa_phase1<<<BB, 256, 0, stream>>>(context, query, w, c2q_ws, q2c_ws);
  cqa_phase2<<<BB * (NN / 8), 256, 0, stream>>>(context, c2q_ws, q2c_ws, out);
}
